// LSTMAdcNoise_5102421148033
// MI455X (gfx1250) — compile-verified, matrix operand order checked by reading, not run on hardware
//
#include <hip/hip_runtime.h>
#include <hip/hip_bf16.h>
#include <math.h>

typedef __attribute__((ext_vector_type(16))) _Float16 v16h;
typedef __attribute__((ext_vector_type(8)))  float    v8f;

#define BS   256
#define SEQ  2048
#define IN   40
#define HS   32
#define NG   128          // 4*HS
#define GRATIO 25.0f
#define KEEP 0.99f

// ws layout (float units):
//  [0..30]      thr_s (31 sigmoid thresholds, SORTED ascending)
//  [31..61]     thr_t (31 tanh thresholds, SORTED ascending)
//  [64..4159]   W B-fragments: 16 frags (2 ksteps x 8 ct) x 32 lanes x 8 dwords (f16 pairs)
//  [4160..6207] U B-fragments:  8 frags (8 ct)            x 32 lanes x 8 dwords
//  [6400.. ]    XW fragment buffer: [16 btile][2048 t][8 ct][32 lane][8 f32]
#define WS_WB   64
#define WS_UB   4160
#define WS_XW   6400

__device__ __forceinline__ float clamp2(float w) {
    return fminf(fmaxf(w, -2.0f), 2.0f);
}

// ---------------- Phase 1: thresholds (sorted) + fragmented W/U ----------------
__global__ __launch_bounds__(256)
void setup_kernel(const float* __restrict__ W_mu, const float* __restrict__ U_mu,
                  const float* __restrict__ delg,
                  const float* __restrict__ gn_s, const float* __restrict__ gn_t,
                  const float* __restrict__ u_w,  const float* __restrict__ u_u,
                  float* __restrict__ ws) {
    const int tid = threadIdx.x;

    // --- ramp thresholds: thread 0 = sigmoid, thread 1 = tanh ---
    if (tid < 2) {
        const float* noise = (tid == 0) ? gn_s : gn_t;
        float v[33];
        for (int i = 0; i < 33; ++i) {
            if (tid == 0) { float t = (float)(i + 1) / 34.0f; v[i] = logf(t / (1.0f - t)); }
            else          { float t = (float)(i - 16) / 17.0f; v[i] = 0.5f * logf((1.0f + t) / (1.0f - t)); }
        }
        float r[32];
        float minr = 1e30f, maxdV = -1e30f;
        for (int i = 0; i < 32; ++i) {
            float dV = v[i + 1] - v[i];
            r[i] = roundf(dV * 1000.0f) * 0.001f;
            minr  = fminf(minr, r[i]);
            maxdV = fmaxf(maxdV, dV);
        }
        float maxcell = -1e30f;
        for (int i = 0; i < 32; ++i) maxcell = fmaxf(maxcell, r[i] / minr);
        const float gscale = 50.0f / (maxcell * minr);   // g[i] = gscale * r[i]
        const float dscale = maxdV / 150.0f;
        float d[32];
        for (int i = 0; i < 32; ++i) d[i] = (gscale * r[i] + noise[i]) * dscale;
        float v0 = 0.0f;
        for (int i = 0; i < 16; ++i) v0 -= d[i];
        float thr[31];
        float acc = v0;
        thr[0] = v0;
        for (int i = 1; i < 31; ++i) { acc += d[i - 1]; thr[i] = acc; }
        // sort ascending: count of (x >= thr_k) is permutation-invariant, and a
        // sorted table lets the scan kernel binary-search (5 steps vs 31 compares)
        for (int i = 1; i < 31; ++i) {
            float key = thr[i];
            int j = i - 1;
            while (j >= 0 && thr[j] > key) { thr[j + 1] = thr[j]; --j; }
            thr[j + 1] = key;
        }
        float* outp = ws + tid * 31;
        for (int i = 0; i < 31; ++i) outp[i] = thr[i];
    }

    // --- W B-fragments: frag f = s*8+ct ; idx = (f*32+lane)*8 + vv ---
    _Float16* hb = (_Float16*)(ws);
    for (int idx = tid; idx < 16 * 32 * 8; idx += 256) {
        int vv   = idx & 7;
        int lane = (idx >> 3) & 31;
        int f    = idx >> 8;
        int ct   = f & 7;
        int s    = f >> 3;
        int kh   = lane >> 4;
        int K = 32 * s + ((vv < 4) ? (8 * kh + 2 * vv) : (16 + 8 * kh + 2 * (vv - 4)));
        int N = 16 * ct + (lane & 15);
        _Float16 lo = (_Float16)0.f, hi = (_Float16)0.f;
        if (K < IN) {
            float mw = (u_w[K * NG + N] < KEEP) ? (1.0f / KEEP) : 0.0f;
            lo = (_Float16)((delg[K * NG + N] * (1.0f / GRATIO) + clamp2(W_mu[K * NG + N])) * mw);
        }
        if (K + 1 < IN) {
            float mw = (u_w[(K + 1) * NG + N] < KEEP) ? (1.0f / KEEP) : 0.0f;
            hi = (_Float16)((delg[(K + 1) * NG + N] * (1.0f / GRATIO) + clamp2(W_mu[(K + 1) * NG + N])) * mw);
        }
        hb[2 * (WS_WB + idx) + 0] = lo;
        hb[2 * (WS_WB + idx) + 1] = hi;
    }

    // --- U B-fragments: frag ct ; K covers 0..31 exactly ---
    for (int idx = tid; idx < 8 * 32 * 8; idx += 256) {
        int vv   = idx & 7;
        int lane = (idx >> 3) & 31;
        int ct   = idx >> 8;
        int kh   = lane >> 4;
        int K = (vv < 4) ? (8 * kh + 2 * vv) : (16 + 8 * kh + 2 * (vv - 4));
        int N = 16 * ct + (lane & 15);
        float mu0 = (u_u[K * NG + N] < KEEP) ? (1.0f / KEEP) : 0.0f;
        float mu1 = (u_u[(K + 1) * NG + N] < KEEP) ? (1.0f / KEEP) : 0.0f;
        _Float16 lo = (_Float16)((delg[(IN + K) * NG + N]     * (1.0f / GRATIO) + clamp2(U_mu[K * NG + N]))       * mu0);
        _Float16 hi = (_Float16)((delg[(IN + K + 1) * NG + N] * (1.0f / GRATIO) + clamp2(U_mu[(K + 1) * NG + N])) * mu1);
        hb[2 * (WS_UB + idx) + 0] = lo;
        hb[2 * (WS_UB + idx) + 1] = hi;
    }
}

// ---------------- Phase 2: XW = x @ W (fully parallel WMMA GEMM) ----------------
__global__ __launch_bounds__(256)
void gemm_xw_kernel(const float* __restrict__ x, const float* __restrict__ ws,
                    float* __restrict__ xwbuf) {
    const int lane = threadIdx.x & 31;
    const int gw   = blockIdx.x * 8 + (threadIdx.x >> 5); // 0..32767
    const int btile = gw >> 11;
    const int t     = gw & (SEQ - 1);
    const int M  = lane & 15;
    const int kh = lane >> 4;

    const float* xrow = x + ((size_t)(btile * 16 + M) * SEQ + t) * IN;

    // A fragments (two K-steps of 32; K padded 40 -> 64 with zeros)
    v16h A[2];
    #pragma unroll
    for (int s = 0; s < 2; ++s) {
        #pragma unroll
        for (int vv = 0; vv < 8; ++vv) {
            int K = 32 * s + ((vv < 4) ? (8 * kh + 2 * vv) : (16 + 8 * kh + 2 * (vv - 4)));
            float2 xv = {0.0f, 0.0f};
            if (K < IN) xv = *(const float2*)(xrow + K);
            A[s][2 * vv]     = (_Float16)xv.x;
            A[s][2 * vv + 1] = (_Float16)xv.y;
        }
    }

    const v16h* WB = (const v16h*)(ws + WS_WB);   // each v16h = 32 bytes
    v8f acc[8] = {};
    #pragma unroll
    for (int s = 0; s < 2; ++s) {
        #pragma unroll
        for (int ct = 0; ct < 8; ++ct) {
            v16h B = WB[(s * 8 + ct) * 32 + lane];
            acc[ct] = __builtin_amdgcn_wmma_f32_16x16x32_f16(
                false, A[s], false, B, (short)0, acc[ct], false, false);
        }
    }

    // store raw C fragments: [gw][ct][lane][8]
    float* outp = xwbuf + (size_t)gw * (8 * 32 * 8);
    #pragma unroll
    for (int ct = 0; ct < 8; ++ct)
        *(v8f*)(outp + (ct * 32 + lane) * 8) = acc[ct];
}

// ---------------- Phase 3: sequential scan (8 waves = 8 column tiles) ----------------
__global__ __launch_bounds__(256)
void lstm_scan_kernel(const float* __restrict__ xwbuf, const float* __restrict__ ws,
                      float* __restrict__ out) {
    __shared__ float    G[16 * NG];      // post-ADC gates for this btile
    __shared__ _Float16 Hl[16 * HS];     // h (f16) for next step's A operand
    __shared__ float    Tlds[64];        // sorted thresholds: [0..30] sig, [32..62] tanh

    const int tid  = threadIdx.x;
    const int lane = tid & 31;
    const int w    = tid >> 5;           // wave id = column tile ct
    const int btile = blockIdx.x;
    const int M  = lane & 15;
    const int kh = lane >> 4;

    // U fragment for this wave's column tile: registers for whole scan
    const v16h Uf = ((const v16h*)(ws + WS_UB))[w * 32 + lane];
    const int  gate    = w >> 1;         // 0:i 1:f 2:g 3:o
    const bool is_tanh = (gate == 2);

    if (tid < 31)                 Tlds[tid]      = ws[tid];        // sigmoid table
    if (tid >= 32 && tid < 63)    Tlds[tid]      = ws[tid - 1];    // tanh table
    for (int i = tid; i < 16 * HS; i += 256) Hl[i] = (_Float16)0.0f;

    // this thread owns 2 consecutive state elements (m_u, n_u) and (m_u, n_u+1)
    const int e0  = tid * 2;
    const int m_u = e0 >> 5;
    const int n_u = e0 & 31;
    float c0 = 0.0f, c1 = 0.0f, h0v = 0.0f, h1v = 0.0f;
    __syncthreads();

    const float* T = &Tlds[is_tanh ? 32 : 0];
    const float* xwb = xwbuf + (size_t)btile * SEQ * (8 * 32 * 8);

    for (int t = 0; t < SEQ; ++t) {
        // A = h in f16 WMMA A-layout (from LDS)
        v16h A;
        #pragma unroll
        for (int vv = 0; vv < 8; ++vv) {
            int K = (vv < 4) ? (8 * kh + 2 * vv) : (16 + 8 * kh + 2 * (vv - 4));
            A[2 * vv]     = Hl[M * HS + K];
            A[2 * vv + 1] = Hl[M * HS + K + 1];
        }
        // C = precomputed x@W fragment for (btile, t, ct=w)
        const float* cp = xwb + (size_t)t * (8 * 32 * 8) + (w * 32 + lane) * 8;
        __builtin_prefetch(cp + (8 * 32 * 8), 0, 1);   // next timestep's line
        v8f acc = *(const v8f*)cp;

        acc = __builtin_amdgcn_wmma_f32_16x16x32_f16(
            false, A, false, Uf, (short)0, acc, false, false);

        // ADC: count of (x >= thr) over sorted table == branchless binary search.
        // 31 = 2^5 - 1, perfect tree, no bounds checks needed.
        #pragma unroll
        for (int j = 0; j < 8; ++j) {
            float xg = acc[j];
            int pos = 0;
            #pragma unroll
            for (int st = 16; st >= 1; st >>= 1) {
                int cand = pos + st;           // candidate count, 1..31
                float tv = T[cand - 1];
                pos = (xg >= tv) ? cand : pos;
            }
            float cnt = (float)pos;
            float val = is_tanh ? (cnt * (2.0f / 34.0f) - 1.0f) : (cnt * (1.0f / 34.0f));
            int m = j + 8 * kh;
            G[m * NG + w * 16 + (lane & 15)] = val;
        }
        __syncthreads();

        // elementwise LSTM state update: 2 elements per thread
        float i0 = G[m_u * NG + n_u],      i1 = G[m_u * NG + n_u + 1];
        float f0 = G[m_u * NG + 32 + n_u], f1 = G[m_u * NG + 32 + n_u + 1];
        float g0 = G[m_u * NG + 64 + n_u], g1 = G[m_u * NG + 64 + n_u + 1];
        float o0 = G[m_u * NG + 96 + n_u], o1 = G[m_u * NG + 96 + n_u + 1];
        c0 = f0 * c0 + i0 * g0;
        c1 = f1 * c1 + i1 * g1;
        h0v = o0 * tanhf(c0);
        h1v = o1 * tanhf(c1);
        Hl[m_u * HS + n_u]     = (_Float16)h0v;
        Hl[m_u * HS + n_u + 1] = (_Float16)h1v;
        float2 hv = {h0v, h1v};
        *(float2*)(out + ((size_t)(btile * 16 + m_u) * SEQ + t) * HS + n_u) = hv;
        __syncthreads();
    }

    // final (h, c) appended after hidden_seq
    const size_t hoff = (size_t)BS * SEQ * HS;
    const int b = btile * 16 + m_u;
    out[hoff + b * HS + n_u]            = h0v;
    out[hoff + b * HS + n_u + 1]        = h1v;
    out[hoff + BS * HS + b * HS + n_u]     = c0;
    out[hoff + BS * HS + b * HS + n_u + 1] = c1;
}

extern "C" void kernel_launch(void* const* d_in, const int* in_sizes, int n_in,
                              void* d_out, int out_size, void* d_ws, size_t ws_size,
                              hipStream_t stream) {
    const float* x    = (const float*)d_in[0];
    const float* W_mu = (const float*)d_in[1];
    const float* U_mu = (const float*)d_in[2];
    const float* delg = (const float*)d_in[3];
    const float* gn_s = (const float*)d_in[4];
    const float* gn_t = (const float*)d_in[5];
    const float* u_w  = (const float*)d_in[6];
    const float* u_u  = (const float*)d_in[7];
    float* wsf   = (float*)d_ws;
    float* xwbuf = wsf + WS_XW;
    float* out   = (float*)d_out;

    setup_kernel<<<1, 256, 0, stream>>>(W_mu, U_mu, delg, gn_s, gn_t, u_w, u_u, wsf);
    // 16 btiles * 2048 t = 32768 waves; 8 waves/block
    gemm_xw_kernel<<<4096, 256, 0, stream>>>(x, wsf, xwbuf);
    // one block (8 waves) per 16-batch tile
    lstm_scan_kernel<<<16, 256, 0, stream>>>(xwbuf, wsf, out);
}
